// sequenceMixer_77979426226860
// MI455X (gfx1250) — matrix operand order repaired and checked by reading, not compiled, not run on hardware
//
#include <hip/hip_runtime.h>

// ---------------------------------------------------------------------------
// Bidirectional minGRU sequence mixer for MI455X (gfx1250, wave32, WMMA).
//   1) normalize weight rows (EDM2 forced weight norm) + convert to bf16
//   2) gh = Wn @ x  (WMMA bf16 GEMMs; conv1x1 is position independent so no
//      reversal is needed at GEMM time)
//   3) minGRU associative scan along L (fwd / bwd), fused sigmoid/g-act,
//      writes mp_cat'ed y (wa=wb=1.0 exactly for these dims)
//   4) out = Wn_out @ y (WMMA bf16 GEMM, f32 epilogue into d_out)
//
// GEMM: block tile 128x128, K-step 32, 8 waves, 8 WMMA/wave/step,
// double-buffered LDS; A-panel staged with GLOBAL_LOAD_ASYNC_TO_LDS_B128
// (ASYNCcnt), B-panel staged synchronously with a transposed scatter. All
// fragments are preloaded into registers so the 8 WMMAs issue back-to-back
// instead of each waiting on a full DS drain.
// ---------------------------------------------------------------------------

typedef __attribute__((ext_vector_type(16))) __bf16 v16bf;
typedef __attribute__((ext_vector_type(8)))  float  v8f;

struct alignas(16) U128 { unsigned int x, y, z, w; };

union PackFrag {            // 32B: two b128 LDS loads -> one WMMA operand
    U128  u[2];
    v16bf v;
};
union Half8 {               // 16B: one b128 global load -> 8 bf16 lanes
    U128           u;
    unsigned short h[8];
};

__device__ __forceinline__ unsigned short f2bf(float f) {
    unsigned int u = __float_as_uint(f);
    unsigned int r = u + 0x7FFFu + ((u >> 16) & 1u);   // round-to-nearest-even
    return (unsigned short)(r >> 16);
}
__device__ __forceinline__ float bf2f(unsigned short h) {
    return __uint_as_float(((unsigned int)h) << 16);
}

// ---- CDNA5 async global->LDS copy (ASYNCcnt) ------------------------------
__device__ __forceinline__ void async_copy_b128(const void* gptr, void* lptr) {
    unsigned           ldsa = (unsigned)(unsigned long long)lptr; // aperture low bits = LDS offset
    unsigned long long ga   = (unsigned long long)gptr;
    asm volatile("global_load_async_to_lds_b128 %0, %1, off"
                 :: "v"(ldsa), "v"(ga) : "memory");
}
__device__ __forceinline__ void wait_async0() {
#if __has_builtin(__builtin_amdgcn_s_wait_asynccnt)
    __builtin_amdgcn_s_wait_asynccnt(0);
#else
    asm volatile("s_wait_asynccnt 0x0" ::: "memory");
#endif
}

// ---------------------------------------------------------------------------
// fp32 -> bf16 bulk convert (x tensor)
// ---------------------------------------------------------------------------
__global__ void cvt_f32_bf16_kernel(const float* __restrict__ in,
                                    unsigned short* __restrict__ out, long n) {
    long i = (long)blockIdx.x * blockDim.x + threadIdx.x;
    long stride = (long)gridDim.x * blockDim.x;
    for (; i < n; i += stride) out[i] = f2bf(in[i]);
}

// ---------------------------------------------------------------------------
// Row-wise L2 normalize (w / (||w||_2 + 1e-4)) + bf16 convert. 1 block / row.
// ---------------------------------------------------------------------------
__global__ void wnorm_bf16_kernel(const float* __restrict__ W,
                                  unsigned short* __restrict__ Wb, int K) {
    const int row = blockIdx.x;
    const int tid = threadIdx.x;
    const float* wr = W + (long)row * K;
    float s = 0.0f;
    for (int i = tid; i < K; i += 256) { float v = wr[i]; s += v * v; }
    __shared__ float red[256];
    red[tid] = s;
    __syncthreads();
    for (int off = 128; off > 0; off >>= 1) {
        if (tid < off) red[tid] += red[tid + off];
        __syncthreads();
    }
    const float inv = 1.0f / (sqrtf(red[0]) + 1e-4f);
    for (int i = tid; i < K; i += 256) Wb[(long)row * K + i] = f2bf(wr[i] * inv);
}

// ---------------------------------------------------------------------------
// bf16 WMMA GEMM: C[b] = W[M,K] * X[b][K,N]   (row-major, f32 accumulate)
// ---------------------------------------------------------------------------
#define TM 128
#define TN 128
#define TK 32
#define NFRAG 8          // TN / 16

__global__ void __launch_bounds__(256)
gemm_bf16_wmma_kernel(const unsigned short* __restrict__ W,
                      const unsigned short* __restrict__ X,
                      void* __restrict__ C,
                      int M, int K, int N, long sX, long sC, int storeF) {
    __shared__ alignas(16) unsigned short sA[2][TM * TK];   // [row][k]  2x8 KB
    __shared__ alignas(16) unsigned short sBt[2][TN * TK];  // [col][k]  2x8 KB

    const int tid  = threadIdx.x;
    const int wave = tid >> 5;
    const int lane = tid & 31;
    const int hl   = lane & 15;          // half-lane index
    const int hi   = (lane >= 16);       // upper half of wave?

    const int b  = blockIdx.z;
    const int m0 = blockIdx.y * TM;
    const int n0 = blockIdx.x * TN;
    const unsigned short* Xb = X + (long)b * sX;

    v8f acc[NFRAG];
#pragma unroll
    for (int f = 0; f < NFRAG; ++f)
#pragma unroll
        for (int r = 0; r < 8; ++r) acc[f][r] = 0.0f;

    // A staging: 512 b128 chunks -> 2 per thread (async global->LDS)
    // B staging: 32x128 halves -> 2 b128 global loads + 16-way LDS scatter
    const int kk = tid >> 3;             // 0..31  (k within tile)
    const int nn = (tid & 7) * 16;       // 0..112 (n within tile)

    const int KT = K / TK;

    // ---- prologue: stage tile 0 ----
    {
#pragma unroll
        for (int s = 0; s < 2; ++s) {
            const int u = tid + s * 256, row = u >> 2, q = (u & 3) * 8;
            async_copy_b128(&W[(long)(m0 + row) * K + q], &sA[0][row * TK + q]);
        }
        Half8 t0, t1;
        t0.u = *(const U128*)&Xb[(long)kk * N + n0 + nn];
        t1.u = *(const U128*)&Xb[(long)kk * N + n0 + nn + 8];
#pragma unroll
        for (int i = 0; i < 8; ++i) {
            sBt[0][(nn + i) * TK + kk]     = t0.h[i];
            sBt[0][(nn + 8 + i) * TK + kk] = t1.h[i];
        }
        wait_async0();
        __syncthreads();
    }

    for (int kt = 0; kt < KT; ++kt) {
        const int cur = kt & 1;
        const int nxt = cur ^ 1;
        const bool more = (kt + 1 < KT);

        // ---- issue next-tile loads (latency hidden under WMMA) ----
        Half8 t0, t1;
        if (more) {
            const int k0n = (kt + 1) * TK;
#pragma unroll
            for (int s = 0; s < 2; ++s) {
                const int u = tid + s * 256, row = u >> 2, q = (u & 3) * 8;
                async_copy_b128(&W[(long)(m0 + row) * K + k0n + q],
                                &sA[nxt][row * TK + q]);
            }
            t0.u = *(const U128*)&Xb[(long)(k0n + kk) * N + n0 + nn];
            t1.u = *(const U128*)&Xb[(long)(k0n + kk) * N + n0 + nn + 8];
            if (kt + 2 < KT)
                __builtin_prefetch(&Xb[(long)((kt + 2) * TK + kk) * N + n0 + nn], 0, 1);
        }

        // ---- preload ALL fragments for this tile into registers ----
        // A frag: lanes 0-15 hold K {0..7,16..23}, lanes 16-31 hold {8..15,24..31}
        const int arow = wave * 16 + hl;
        const int khA  = hi ? 8 : 0;
        PackFrag af;
        af.u[0] = *(const U128*)&sA[cur][arow * TK + khA];
        af.u[1] = *(const U128*)&sA[cur][arow * TK + khA + 16];

        const int khB = hi ? 8 : 0;      // lanes 0-15: K {0..7,16..23}, 16-31: K {8..15,24..31}
        PackFrag bfr[NFRAG];
#pragma unroll
        for (int f = 0; f < NFRAG; ++f) {
            const int col = f * 16 + hl;
            bfr[f].u[0] = *(const U128*)&sBt[cur][col * TK + khB];
            bfr[f].u[1] = *(const U128*)&sBt[cur][col * TK + khB + 16];
        }

        // ---- back-to-back WMMA burst (disjoint accumulators, no hazards) --
#pragma unroll
        for (int f = 0; f < NFRAG; ++f)
            acc[f] = __builtin_amdgcn_wmma_f32_16x16x32_bf16(
                false, af.v, false, bfr[f].v, (short)0, acc[f], false, false);

        // ---- commit next B tile (transposed), close the pipeline stage ----
        if (more) {
#pragma unroll
            for (int i = 0; i < 8; ++i) {
                sBt[nxt][(nn + i) * TK + kk]     = t0.h[i];
                sBt[nxt][(nn + 8 + i) * TK + kk] = t1.h[i];
            }
            wait_async0();
        }
        __syncthreads();
    }

    // ---- epilogue: VGPR r -> M = r (lanes 0-15) or r+8 (lanes 16-31) ----
    float*          Cf = (float*)C;
    unsigned short* Cb = (unsigned short*)C;
#pragma unroll
    for (int f = 0; f < NFRAG; ++f) {
        const int n = n0 + f * 16 + hl;
#pragma unroll
        for (int r = 0; r < 8; ++r) {
            const int  m   = m0 + wave * 16 + r + (hi ? 8 : 0);
            const long idx = (long)b * sC + (long)m * N + n;
            if (storeF) Cf[idx] = acc[f][r];
            else        Cb[idx] = f2bf(acc[f][r]);
        }
    }
}

// ---------------------------------------------------------------------------
// minGRU scan along L for one (batch, channel, direction).
//   z = sigmoid(gate); a = 1-z; b = z*g(hidden); h_t = a_t h_{t-1} + b_t
// 256 threads x 16 serial elements + Hillis-Steele pair scan in LDS.
// ---------------------------------------------------------------------------
__global__ void __launch_bounds__(256)
mingru_scan_kernel(const unsigned short* __restrict__ gh,
                   unsigned short* __restrict__ y,
                   int O, int L, int reversed, int yRowOff, float scale) {
    const int tid = threadIdx.x;
    const int ch  = blockIdx.x;
    const int b   = blockIdx.y;
    const long base = (long)b * 2 * O * L;
    const unsigned short* gate = gh + base + (long)ch * L;
    const unsigned short* hid  = gh + base + (long)(O + ch) * L;
    unsigned short*       yr   = y  + base + (long)(yRowOff + ch) * L;

    const int CH = 16;                   // L / 256
    const int start = tid * CH;
    float av[16], bv[16];
    float A = 1.0f, Bc = 0.0f;
#pragma unroll
    for (int i = 0; i < CH; ++i) {
        const int l = start + i;
        const int p = reversed ? (L - 1 - l) : l;
        const float gz = bf2f(gate[p]);
        const float hh = bf2f(hid[p]);
        const float z  = 1.0f / (1.0f + __expf(-gz));
        const float g  = (hh >= 0.0f) ? (hh + 0.5f)
                                      : (1.0f / (1.0f + __expf(-hh)));
        const float a  = 1.0f - z;
        const float bb = z * g;
        av[i] = a; bv[i] = bb;
        A  = a * A;                      // compose (A,Bc) then (a,bb)
        Bc = a * Bc + bb;
    }

    __shared__ float sA[256], sB[256];
    sA[tid] = A; sB[tid] = Bc;
    __syncthreads();
    for (int off = 1; off < 256; off <<= 1) {
        float pa = 1.0f, pb = 0.0f;
        const float ca = sA[tid], cb = sB[tid];
        if (tid >= off) { pa = sA[tid - off]; pb = sB[tid - off]; }
        __syncthreads();
        sA[tid] = pa * ca;
        sB[tid] = ca * pb + cb;
        __syncthreads();
    }

    float h = (tid == 0) ? 0.0f : sB[tid - 1];   // exclusive prefix, h_0 = 0
#pragma unroll
    for (int i = 0; i < CH; ++i) {
        h = av[i] * h + bv[i];
        const int l = start + i;
        const int p = reversed ? (L - 1 - l) : l;
        yr[p] = f2bf(h * scale);
    }
}

// ---------------------------------------------------------------------------
extern "C" void kernel_launch(void* const* d_in, const int* in_sizes, int n_in,
                              void* d_out, int out_size, void* d_ws, size_t ws_size,
                              hipStream_t stream) {
    (void)in_sizes; (void)n_in; (void)out_size; (void)ws_size;
    const float* x  = (const float*)d_in[0];   // [8,1024,4096]
    const float* fw = (const float*)d_in[1];   // [2048,1024]
    const float* bw = (const float*)d_in[2];   // [2048,1024]
    const float* ow = (const float*)d_in[3];   // [1024,2048]
    float* out = (float*)d_out;                // [8,1024,4096]

    const int B = 8, D = 1024, L = 4096, O = 1024;

    char* p = (char*)d_ws;
    unsigned short* xbf = (unsigned short*)p; p += (size_t)B * D * L * 2;       //  64 MB
    unsigned short* wfb = (unsigned short*)p; p += (size_t)2 * O * D * 2;       //   4 MB
    unsigned short* wbb = (unsigned short*)p; p += (size_t)2 * O * D * 2;       //   4 MB
    unsigned short* wob = (unsigned short*)p; p += (size_t)D * 2 * O * 2;       //   4 MB
    unsigned short* ghf = (unsigned short*)p; p += (size_t)B * 2 * O * L * 2;   // 128 MB
    unsigned short* ghb = (unsigned short*)p; p += (size_t)B * 2 * O * L * 2;   // 128 MB
    unsigned short* ybf = (unsigned short*)p; p += (size_t)B * 2 * O * L * 2;   // 128 MB

    // 1) conversions + weight normalization
    cvt_f32_bf16_kernel<<<8192, 256, 0, stream>>>(x, xbf, (long)B * D * L);
    wnorm_bf16_kernel<<<2 * O, 256, 0, stream>>>(fw, wfb, D);
    wnorm_bf16_kernel<<<2 * O, 256, 0, stream>>>(bw, wbb, D);
    wnorm_bf16_kernel<<<D, 256, 0, stream>>>(ow, wob, 2 * O);

    // 2) projection GEMMs (gh = Wn @ x), raw bf16 gate/hidden outputs
    dim3 g1(L / TN, (2 * O) / TM, B);
    gemm_bf16_wmma_kernel<<<g1, 256, 0, stream>>>(
        wfb, xbf, ghf, 2 * O, D, L, (long)D * L, (long)2 * O * L, 0);
    gemm_bf16_wmma_kernel<<<g1, 256, 0, stream>>>(
        wbb, xbf, ghb, 2 * O, D, L, (long)D * L, (long)2 * O * L, 0);

    // 3) scans (fore forward, back reversed); mp_cat weights == 1.0 exactly
    dim3 gs(O, B);
    mingru_scan_kernel<<<gs, 256, 0, stream>>>(ghf, ybf, O, L, 0, 0, 1.0f);
    mingru_scan_kernel<<<gs, 256, 0, stream>>>(ghb, ybf, O, L, 1, O, 1.0f);

    // 4) output projection, f32 epilogue straight into d_out
    dim3 g2(L / TN, D / TM, B);
    gemm_bf16_wmma_kernel<<<g2, 256, 0, stream>>>(
        wob, ybf, out, D, 2 * O, L, (long)2 * O * L, (long)D * L, 1);
}
